// TimeGanNet_70033736729107
// MI455X (gfx1250) — matrix operand order repaired and checked by reading, not compiled, not run on hardware
//
#include <hip/hip_runtime.h>

// CDNA5 / gfx1250, wave32. Fused GRU + FC head for TimeGAN.
// One workgroup owns 16 batch rows for all 256 timesteps; hidden state lives in
// per-thread registers, weights live in register-resident WMMA B fragments.
// All matmuls use v_wmma_f32_16x16x32_f16 (f32 accumulate); activations use the
// gfx1250 transcendental units. All global accesses index directly off kernel
// args so they lower to global_* (not flat_*) ops.

typedef __attribute__((ext_vector_type(16))) _Float16 v16h;
typedef __attribute__((ext_vector_type(8)))  _Float16 v8h;
typedef __attribute__((ext_vector_type(8)))  float    v8f;

#define B_TOT   4096
#define T_SEQ   256
#define IN_SZ   24
#define OUT_SZ  24
#define HID     40
#define G3      120      // 3*HID

#define BT       16      // batch rows per workgroup
#define NTHREADS 128     // 4 waves (wave32)

#ifndef __has_builtin
#define __has_builtin(x) 0
#endif

#define LOG2E 1.44269504088896f

__device__ __forceinline__ float sigm(float v) {
    // 1/(1+e^-v) via v_exp_f32 (base-2) + v_rcp_f32 — no IEEE divide expansion
    float e = __builtin_amdgcn_exp2f(v * -LOG2E);
    return __builtin_amdgcn_rcpf(1.0f + e);
}

__device__ __forceinline__ float fast_tanh(float x) {
#if __has_builtin(__builtin_amdgcn_tanhf)
    return __builtin_amdgcn_tanhf(x);                 // v_tanh_f32
#else
    // tanh(x) = 1 - 2/(e^{2x}+1)
    float e = __builtin_amdgcn_exp2f(x * (2.0f * LOG2E));
    return 1.0f - 2.0f * __builtin_amdgcn_rcpf(e + 1.0f);
#endif
}

// A fragment (16x32 f16, M=batch rows), ISA layout:
// lane l<16: M=l, K = kb..kb+7 and kb+16..kb+23; lane l>=16: M=l-16, kb += 8
__device__ __forceinline__ v16h load_a_frag(const _Float16* base, int ld, int kbase, int lane) {
    int m  = lane & 15;
    int kb = kbase + ((lane & 16) ? 8 : 0);
    v8h lo = *(const v8h*)(base + m * ld + kb);
    v8h hi = *(const v8h*)(base + m * ld + kb + 16);
    v16h a;
#pragma unroll
    for (int i = 0; i < 8; ++i) { a[i] = lo[i]; a[8 + i] = hi[i]; }
    return a;
}

// B fragment (32x16 f16, from row-major W[k][n] in LDS):
// lane l<16: N=nbase+l, K=kbase..kbase+7 and kbase+16..kbase+23; lane>=16: K += 8
__device__ __forceinline__ v16h load_b_frag(const _Float16* W, int ldn, int kbase, int nbase, int lane) {
    int n  = nbase + (lane & 15);
    int k0 = kbase + ((lane & 16) ? 8 : 0);
    v16h b;
#pragma unroll
    for (int v = 0; v < 8; ++v) {
        b[v]     = W[(k0 + v)      * ldn + n];
        b[8 + v] = W[(k0 + 16 + v) * ldn + n];
    }
    return b;
}

// D (16x16 f32): VGPR v -> row v (lanes 0-15) / row v+8 (lanes 16-31), col = lane&15
__device__ __forceinline__ void store_d(float* base, int ld, int nt, int lane, v8f acc) {
    int col = nt * 16 + (lane & 15);
    int r0  = (lane & 16) ? 8 : 0;
#pragma unroll
    for (int v = 0; v < 8; ++v) base[(r0 + v) * ld + col] = acc[v];
}

__global__ __launch_bounds__(NTHREADS)
void timegan_gru_fused(const float* __restrict__ x,
                       const float* __restrict__ W_ih,
                       const float* __restrict__ W_hh,
                       const float* __restrict__ b_ih,
                       const float* __restrict__ b_hh,
                       const float* __restrict__ W_fc,
                       const float* __restrict__ b_fc,
                       float* __restrict__ out) {
    // Overlay: weight staging (28 KB, init only) aliases per-step gate buffers (16 KB).
    __shared__ __align__(16) char sOverlay[28 * 1024];
    _Float16* sWih = (_Float16*)sOverlay;                    // [32][128] f16 (K 24->32)
    _Float16* sWhh = (_Float16*)(sOverlay + 8192);           // [64][128] f16 (K 40->64)
    _Float16* sWfc = (_Float16*)(sOverlay + 8192 + 16384);   // [64][32]  f16
    float*    sXg  = (float*)sOverlay;                       // [16][128] x-projection
    float*    sHg  = (float*)(sOverlay + 8192);              // [16][128] h-projection

    __shared__ __align__(16) _Float16 sX16[BT * 32];         // x_t tile, f16, K-padded
    __shared__ __align__(16) _Float16 sH16[BT * 64];         // h, f16, K-padded 40->64
    __shared__ float sBih[G3], sBhh[G3], sBfc[OUT_SZ];

    const int tid  = threadIdx.x;
    const int lane = tid & 31;
    const int w    = tid >> 5;          // wave 0..3
    const int b0   = blockIdx.x * BT;

    // ---- stage weights (f32 -> f16, zero-padded) ----
    for (int i = tid; i < 32 * 128; i += NTHREADS) {
        int n = i & 127, k = i >> 7;
        sWih[i] = (_Float16)((k < IN_SZ && n < G3) ? W_ih[n * IN_SZ + k] : 0.0f);
    }
    for (int i = tid; i < 64 * 128; i += NTHREADS) {
        int n = i & 127, k = i >> 7;
        sWhh[i] = (_Float16)((k < HID && n < G3) ? W_hh[n * HID + k] : 0.0f);
    }
    for (int i = tid; i < 64 * 32; i += NTHREADS) {
        int n = i & 31, k = i >> 5;
        sWfc[i] = (_Float16)((k < HID && n < OUT_SZ) ? W_fc[n * HID + k] : 0.0f);
    }
    for (int i = tid; i < G3; i += NTHREADS) { sBih[i] = b_ih[i]; sBhh[i] = b_hh[i]; }
    if (tid < OUT_SZ) sBfc[tid] = b_fc[tid];
    for (int i = tid; i < BT * 64; i += NTHREADS) sH16[i] = (_Float16)0.0f;
    for (int i = tid; i < BT * 32; i += NTHREADS) sX16[i] = (_Float16)0.0f;   // cols>=24 stay 0
    __syncthreads();

    // ---- register-resident B fragments (loop-invariant weights) ----
    const int nt0 = 2 * w;                              // this wave owns N-tiles nt0, nt0+1
    v16h Bih0  = load_b_frag(sWih, 128, 0,  nt0 * 16,        lane);
    v16h Bih1  = load_b_frag(sWih, 128, 0,  (nt0 + 1) * 16,  lane);
    v16h Bhh00 = load_b_frag(sWhh, 128, 0,  nt0 * 16,        lane);
    v16h Bhh01 = load_b_frag(sWhh, 128, 32, nt0 * 16,        lane);
    v16h Bhh10 = load_b_frag(sWhh, 128, 0,  (nt0 + 1) * 16,  lane);
    v16h Bhh11 = load_b_frag(sWhh, 128, 32, (nt0 + 1) * 16,  lane);
    v16h Bfc0 = {}, Bfc1 = {};
    const bool doFC  = (w < 2);
    const int  fc_nt = w;                               // waves 0,1 own the two FC N-tiles
    if (doFC) {
        Bfc0 = load_b_frag(sWfc, 32, 0,  fc_nt * 16, lane);
        Bfc1 = load_b_frag(sWfc, 32, 32, fc_nt * 16, lane);
    }

    // ---- hoist loop-invariant addressing / biases into registers ----
    // Phase A: 3 x-elements per thread (integer offsets, NOT pointers, so the
    // accesses stay anchored to the global kernel arg -> global_load_b32).
    int    xIdx[3];
    size_t xOff[3];
#pragma unroll
    for (int p = 0; p < 3; ++p) {
        int i = tid + p * NTHREADS;                     // 0..383 (= 16*24)
        int row = i / IN_SZ, col = i % IN_SZ;
        xIdx[p] = row * 32 + col;
        xOff[p] = (size_t)(b0 + row) * T_SEQ * IN_SZ + col;
    }
    // Phase C: 5 gate cells per thread; h lives in registers
    int    gIdx[5], hIdx[5];
    size_t hoIdx[5];
    float  br[5], bz[5], bxn[5], bhn[5], hreg[5];
    const size_t HBASE = (size_t)B_TOT * T_SEQ * OUT_SZ;
#pragma unroll
    for (int p = 0; p < 5; ++p) {
        int i = tid + p * NTHREADS;                     // 0..639 (= 16*40)
        int row = i / HID, j = i % HID;
        gIdx[p]  = row * 128 + j;
        hIdx[p]  = row * 64 + j;
        hoIdx[p] = HBASE + (size_t)(b0 + row) * HID + j;
        br[p]  = sBih[j]      + sBhh[j];
        bz[p]  = sBih[40 + j] + sBhh[40 + j];
        bxn[p] = sBih[80 + j];
        bhn[p] = sBhh[80 + j];
        hreg[p] = 0.0f;
    }
    // Phase D: output addressing (integer offset)
    const int  col   = fc_nt * 16 + (lane & 15);
    const bool colOK = doFC && (col < OUT_SZ);
    const int  r0    = (lane & 16) ? 8 : 0;
    const float bb   = colOK ? sBfc[col] : 0.0f;
    size_t outOff = (size_t)(b0 + r0) * T_SEQ * OUT_SZ + col;

    for (int t = 0; t < T_SEQ; ++t) {
        // -- phase A: stage x_t tile; branch-free prefetch of x_{t+1} --
        const size_t pf = (t < T_SEQ - 1) ? (size_t)IN_SZ : 0;  // clamp, no loop split
        float xa[3];
#pragma unroll
        for (int p = 0; p < 3; ++p) {                   // 3 global_load_b32 back-to-back
            xa[p] = x[xOff[p]];
            __builtin_prefetch(&x[xOff[p] + pf], 0, 1); // global_prefetch_b8
        }
#pragma unroll
        for (int p = 0; p < 3; ++p) {
            sX16[xIdx[p]] = (_Float16)xa[p];
            xOff[p] += IN_SZ;
        }
        __syncthreads();   // also orders weight-frag reads vs first sXg write (overlay)

        // -- phase B: gate GEMMs  xg = x_t*W_ih^T ,  hg = h*W_hh^T --
        {
            v16h Ax  = load_a_frag(sX16, 32, 0,  lane);
            v16h Ah0 = load_a_frag(sH16, 64, 0,  lane);
            v16h Ah1 = load_a_frag(sH16, 64, 32, lane);

            v8f aX0 = {}, aH0 = {}, aX1 = {}, aH1 = {};
            aX0 = __builtin_amdgcn_wmma_f32_16x16x32_f16(false, Ax,  false, Bih0,  (short)0, aX0, false, false);
            aH0 = __builtin_amdgcn_wmma_f32_16x16x32_f16(false, Ah0, false, Bhh00, (short)0, aH0, false, false);
            aH0 = __builtin_amdgcn_wmma_f32_16x16x32_f16(false, Ah1, false, Bhh01, (short)0, aH0, false, false);
            aX1 = __builtin_amdgcn_wmma_f32_16x16x32_f16(false, Ax,  false, Bih1,  (short)0, aX1, false, false);
            aH1 = __builtin_amdgcn_wmma_f32_16x16x32_f16(false, Ah0, false, Bhh10, (short)0, aH1, false, false);
            aH1 = __builtin_amdgcn_wmma_f32_16x16x32_f16(false, Ah1, false, Bhh11, (short)0, aH1, false, false);

            store_d(sXg, 128, nt0,     lane, aX0);
            store_d(sHg, 128, nt0,     lane, aH0);
            store_d(sXg, 128, nt0 + 1, lane, aX1);
            store_d(sHg, 128, nt0 + 1, lane, aH1);
        }
        __syncthreads();

        // -- phase C: element-wise GRU gates (f32, h in registers) --
#pragma unroll
        for (int p = 0; p < 5; ++p) {
            int g = gIdx[p];
            float r  = sigm(sXg[g]      + sHg[g]      + br[p]);
            float z  = sigm(sXg[g + 40] + sHg[g + 40] + bz[p]);
            float nv = fast_tanh(sXg[g + 80] + bxn[p] + r * (sHg[g + 80] + bhn[p]));
            float hw = z * (hreg[p] - nv) + nv;        // (1-z)*n + z*h
            hreg[p] = hw;
            sH16[hIdx[p]] = (_Float16)hw;
        }
        __syncthreads();

        // -- phase D: FC head  out_t = sigmoid(h * W_fc^T + b_fc) --
        if (doFC) {
            v16h A0 = load_a_frag(sH16, 64, 0,  lane);
            v16h A1 = load_a_frag(sH16, 64, 32, lane);
            v8f acc = {};
            acc = __builtin_amdgcn_wmma_f32_16x16x32_f16(false, A0, false, Bfc0, (short)0, acc, false, false);
            acc = __builtin_amdgcn_wmma_f32_16x16x32_f16(false, A1, false, Bfc1, (short)0, acc, false, false);
            if (colOK) {
#pragma unroll
                for (int v = 0; v < 8; ++v)            // global_store_b32, 64B runs per row
                    out[outOff + (size_t)v * (T_SEQ * OUT_SZ)] = sigm(acc[v] + bb);
            }
            outOff += OUT_SZ;                          // advance to next timestep
        }
        // no barrier needed: next phase-A touches only sX16, re-synced before phase B
    }

    // ---- final hidden state (registers) -> out tail [1, B, HID] ----
#pragma unroll
    for (int p = 0; p < 5; ++p) out[hoIdx[p]] = hreg[p];
}

extern "C" void kernel_launch(void* const* d_in, const int* in_sizes, int n_in,
                              void* d_out, int out_size, void* d_ws, size_t ws_size,
                              hipStream_t stream) {
    const float* x    = (const float*)d_in[0];
    const float* W_ih = (const float*)d_in[1];
    const float* W_hh = (const float*)d_in[2];
    const float* b_ih = (const float*)d_in[3];
    const float* b_hh = (const float*)d_in[4];
    const float* W_fc = (const float*)d_in[5];
    const float* b_fc = (const float*)d_in[6];
    float* out = (float*)d_out;

    dim3 grid(B_TOT / BT);      // 256 workgroups
    dim3 block(NTHREADS);       // 4 waves of 32
    hipLaunchKernelGGL(timegan_gru_fused, grid, block, 0, stream,
                       x, W_ih, W_hh, b_ih, b_hh, W_fc, b_fc, out);
}
